// MultiHeadAttention_80822694576181
// MI455X (gfx1250) — matrix operand order repaired and checked by reading, not compiled, not run on hardware
//
#include <hip/hip_runtime.h>
#include <hip/hip_bf16.h>

// ---------------------------------------------------------------------------
// MHA for MI455X (gfx1250): bf16 WMMA (v_wmma_f32_16x16x32_bf16) everywhere.
// B=2 S=2048 D=768 H=12 N=64.  Compute-bound (~45 GFLOP vs ~35MB HBM traffic).
// This revision enlarges per-wave tiles to raise FLOP/byte at the WMMA inputs:
//   GEMM: 64x64 per wave (16 WMMAs / 8KB per k-step)
//   Attention: 32 queries per wave (16 WMMAs / 8KB per 32-key block)
// ---------------------------------------------------------------------------

typedef __attribute__((ext_vector_type(16))) __bf16 v16bf;
typedef __attribute__((ext_vector_type(8)))  __bf16 v8bf;
typedef __attribute__((ext_vector_type(8)))  float  v8f;

#define BATCH 2
#define SEQ   2048
#define DMODEL 768
#define HEADS 12
#define HDIM  64
#define ROWS  (BATCH * SEQ)          // 4096

__device__ __forceinline__ v8f wmma_bf16(v16bf a, v16bf b, v8f c) {
  // D = A(16x32) * B(32x16) + C(16x16 f32)
  return __builtin_amdgcn_wmma_f32_16x16x32_bf16(
      /*neg_a=*/false, a, /*neg_b=*/false, b,
      /*c_mod=*/(short)0, c, /*reuse_a=*/false, /*reuse_b=*/false);
}

// A-fragment (16x32 bf16): lane row = lane&15; lanes<16 hold K {0..15},
// lanes>=16 hold K {16..31}.  Two contiguous 16B loads per lane.
__device__ __forceinline__ v16bf load_a_frag(const __bf16* __restrict__ row,
                                             int kbase, int grp) {
  v8bf h0 = *(const v8bf*)(row + kbase + grp * 16);
  v8bf h1 = *(const v8bf*)(row + kbase + grp * 16 + 8);
  return __builtin_shufflevector(h0, h1, 0, 1, 2, 3, 4, 5, 6, 7,
                                 8, 9, 10, 11, 12, 13, 14, 15);
}

// ---------------------------------------------------------------------------
// elementwise fp32 -> bf16
__global__ void cvt_f32_bf16(const float* __restrict__ src,
                             __bf16* __restrict__ dst, int n) {
  int i = blockIdx.x * blockDim.x + threadIdx.x;
  if (i < n) dst[i] = (__bf16)src[i];
}

// W[k][n] (f32, KxN) -> Wt[n][k] (bf16, NxK) so B-fragments are contiguous.
__global__ void transpose_f32_bf16(const float* __restrict__ W,
                                   __bf16* __restrict__ Wt, int K, int N) {
  int i = blockIdx.x * blockDim.x + threadIdx.x;
  if (i >= K * N) return;
  int n = i / K, k = i % K;
  Wt[i] = (__bf16)W[(long)k * N + n];
}

// ---------------------------------------------------------------------------
// GEMM: C = A(MxK bf16 row-major) * Bt^T + bias.  Wave computes 64x64 tile.
// mode 0: bf16 out at [b][h][s][64]     (Q, K layout)
// mode 1: bf16 out at [b][h][64][s]     (V transposed)
// mode 2: f32  out row-major MxN        (final projection)
__global__ __launch_bounds__(128) void gemm_bf16_wmma(
    const __bf16* __restrict__ A, const __bf16* __restrict__ Bt,
    const float* __restrict__ bias, __bf16* __restrict__ out_bf,
    float* __restrict__ out_f32, int M, int Ktot, int Ntot, int mode) {
  const int lane = threadIdx.x & 31;
  const int wave = threadIdx.x >> 5;
  const int c = lane & 15;       // output column within 16-tile
  const int g = lane >> 4;       // half-wave group
  const int mbase = (blockIdx.x * 4 + wave) * 64;
  const int ntile = blockIdx.y;  // 64 wide
  if (mbase >= M) return;        // wave-uniform; EXEC stays all-ones

  v8f acc[4][4] = {};            // [m subtile][n subtile]
  for (int k = 0; k < Ktot; k += 32) {
    v16bf a[4];
#pragma unroll
    for (int mt = 0; mt < 4; ++mt)
      a[mt] = load_a_frag(A + (long)(mbase + mt * 16 + c) * Ktot, k, g);
#pragma unroll
    for (int nb = 0; nb < 4; ++nb) {
      const int n = ntile * 64 + nb * 16 + c;
      v16bf b = *(const v16bf*)(Bt + (long)n * Ktot + k + g * 16);
#pragma unroll
      for (int mt = 0; mt < 4; ++mt)
        acc[mt][nb] = wmma_bf16(a[mt], b, acc[mt][nb]);
    }
  }
#pragma unroll
  for (int nb = 0; nb < 4; ++nb) {
    const int n = ntile * 64 + nb * 16 + c;
    const float bv = bias ? bias[n] : 0.0f;
    const int h = n / HDIM, nn = n % HDIM;
#pragma unroll
    for (int mt = 0; mt < 4; ++mt) {
#pragma unroll
      for (int r = 0; r < 8; ++r) {
        const int rg = mbase + mt * 16 + r + 8 * g;  // global row (C layout)
        const float v = acc[mt][nb][r] + bv;
        if (mode == 2) {
          out_f32[(long)rg * Ntot + n] = v;
        } else {
          const int bb = rg / SEQ, s = rg % SEQ;
          if (mode == 0)
            out_bf[(((long)(bb * HEADS + h) * SEQ) + s) * HDIM + nn] = (__bf16)v;
          else
            out_bf[(((long)(bb * HEADS + h) * HDIM) + nn) * SEQ + s] = (__bf16)v;
        }
      }
    }
  }
}

// ---------------------------------------------------------------------------
// Flash attention: one wave owns a 32-query tile (two 16-row C tiles) of one
// head, so K/V fragments are reused across both query sub-tiles.
// Q,K: [b*H][S][64] bf16.  Vt: [b*H][64][S] bf16.  O: [b][s][768] bf16.
__global__ __launch_bounds__(128) void attn_wmma(
    const __bf16* __restrict__ Q, const __bf16* __restrict__ Km,
    const __bf16* __restrict__ Vt, const int* __restrict__ masks,
    __bf16* __restrict__ O) {
  __shared__ float pst[4][2][16][36];  // per-wave, per-qtile P staging
  const int lane = threadIdx.x & 31;
  const int wave = threadIdx.x >> 5;
  const int c = lane & 15;
  const int g = lane >> 4;
  const int wid = blockIdx.x * 4 + wave;     // 1536 waves total
  const int qpair = wid & 63;                // 64 32-query tiles per head
  const int head_lin = wid >> 6;             // b*HEADS + h
  const int b = head_lin / HEADS;
  const int h = head_lin % HEADS;

  const __bf16* qp = Q + (long)head_lin * SEQ * HDIM;
  const __bf16* kp = Km + (long)head_lin * SEQ * HDIM;
  const __bf16* vp = Vt + (long)head_lin * HDIM * SEQ;
  const int* mp = masks + b * SEQ;

  // Q fragments: [q subtile][n chunk], loaded once.
  v16bf aq[2][2];
#pragma unroll
  for (int qt = 0; qt < 2; ++qt) {
    const __bf16* qrow = qp + (long)(qpair * 32 + qt * 16 + c) * HDIM;
    aq[qt][0] = load_a_frag(qrow, 0, g);
    aq[qt][1] = load_a_frag(qrow, 32, g);
  }

  v8f acc[2][4] = {};
  float mrow[2][8], lrow[2][8];
#pragma unroll
  for (int qt = 0; qt < 2; ++qt)
#pragma unroll
    for (int r = 0; r < 8; ++r) { mrow[qt][r] = -1e30f; lrow[qt][r] = 0.0f; }

  for (int j = 0; j < SEQ; j += 32) {
    if (j + 32 < SEQ) __builtin_prefetch(kp + (long)(j + 32 + c) * HDIM, 0, 0);
    // K^T fragments for this 32-key block: [key half][n chunk]
    v16bf bk[2][2];
    bk[0][0] = *(const v16bf*)(kp + (long)(j + c) * HDIM + g * 16);
    bk[0][1] = *(const v16bf*)(kp + (long)(j + c) * HDIM + 32 + g * 16);
    bk[1][0] = *(const v16bf*)(kp + (long)(j + 16 + c) * HDIM + g * 16);
    bk[1][1] = *(const v16bf*)(kp + (long)(j + 16 + c) * HDIM + 32 + g * 16);

    // scores: S[32q x 32k] via 8 WMMAs, K frags shared across q sub-tiles
    v8f s[2][2] = {};
#pragma unroll
    for (int qt = 0; qt < 2; ++qt)
#pragma unroll
      for (int kh = 0; kh < 2; ++kh) {
        s[qt][kh] = wmma_bf16(aq[qt][0], bk[kh][0], s[qt][kh]);
        s[qt][kh] = wmma_bf16(aq[qt][1], bk[kh][1], s[qt][kh]);
      }

    const bool msk0 = (mp[j + c] == 0);
    const bool msk1 = (mp[j + 16 + c] == 0);
#pragma unroll
    for (int qt = 0; qt < 2; ++qt) {
      float corr[8];
#pragma unroll
      for (int r = 0; r < 8; ++r) {
        float e0 = msk0 ? -1e12f : s[qt][0][r] * 0.125f;  // 1/sqrt(64)
        float e1 = msk1 ? -1e12f : s[qt][1][r] * 0.125f;
        float mx = fmaxf(e0, e1);
#pragma unroll
        for (int off = 8; off >= 1; off >>= 1)
          mx = fmaxf(mx, __shfl_xor(mx, off, 32));  // row lives in 16-lane half
        const float mnew = fmaxf(mrow[qt][r], mx);
        const float cr = __expf(mrow[qt][r] - mnew);
        mrow[qt][r] = mnew;
        e0 = __expf(e0 - mnew);
        e1 = __expf(e1 - mnew);
        s[qt][0][r] = e0; s[qt][1][r] = e1;
        float rs = e0 + e1;
#pragma unroll
        for (int off = 8; off >= 1; off >>= 1) rs += __shfl_xor(rs, off, 32);
        lrow[qt][r] = lrow[qt][r] * cr + rs;
        corr[r] = cr;
      }
#pragma unroll
      for (int nb = 0; nb < 4; ++nb)
#pragma unroll
        for (int r = 0; r < 8; ++r) acc[qt][nb][r] *= corr[r];
    }

    // re-layout P: C-form (column per lane) -> A-form (row per lane), via LDS
#pragma unroll
    for (int qt = 0; qt < 2; ++qt)
#pragma unroll
      for (int r = 0; r < 8; ++r) {
        pst[wave][qt][r + 8 * g][c] = s[qt][0][r];
        pst[wave][qt][r + 8 * g][16 + c] = s[qt][1][r];
      }
    __syncthreads();
    v16bf pa[2];
#pragma unroll
    for (int qt = 0; qt < 2; ++qt) {
#pragma unroll
      for (int i = 0; i < 8; ++i)
        pa[qt][i] = (__bf16)pst[wave][qt][c][g * 16 + i];
#pragma unroll
      for (int i = 0; i < 8; ++i)
        pa[qt][8 + i] = (__bf16)pst[wave][qt][c][g * 16 + 8 + i];
    }
    __syncthreads();

    // O += P(16x32) * V(32x16); V frags shared across q sub-tiles
#pragma unroll
    for (int nb = 0; nb < 4; ++nb) {
      v16bf bv = *(const v16bf*)(vp + (long)(nb * 16 + c) * SEQ + j + g * 16);
#pragma unroll
      for (int qt = 0; qt < 2; ++qt)
        acc[qt][nb] = wmma_bf16(pa[qt], bv, acc[qt][nb]);
    }
  }

  // epilogue: normalize and scatter to O[b][s][h*64 + n] (bf16, head concat)
#pragma unroll
  for (int qt = 0; qt < 2; ++qt)
#pragma unroll
    for (int r = 0; r < 8; ++r) {
      const float inv = 1.0f / lrow[qt][r];
      const int s = qpair * 32 + qt * 16 + r + 8 * g;
#pragma unroll
      for (int nb = 0; nb < 4; ++nb)
        O[((long)b * SEQ + s) * DMODEL + h * HDIM + nb * 16 + c] =
            (__bf16)(acc[qt][nb][r] * inv);
    }
}

// ---------------------------------------------------------------------------
extern "C" void kernel_launch(void* const* d_in, const int* in_sizes, int n_in,
                              void* d_out, int out_size, void* d_ws,
                              size_t ws_size, hipStream_t stream) {
  const float* x  = (const float*)d_in[0];
  const int* mk   = (const int*)d_in[1];
  const float* Wq = (const float*)d_in[2];
  const float* bq = (const float*)d_in[3];
  const float* Wk = (const float*)d_in[4];
  const float* bk = (const float*)d_in[5];
  const float* Wv = (const float*)d_in[6];
  const float* bv = (const float*)d_in[7];
  const float* Wo = (const float*)d_in[8];
  const float* bo = (const float*)d_in[9];
  float* out = (float*)d_out;

  char* ws = (char*)d_ws;
  // workspace layout (bytes)
  __bf16* Xbf = (__bf16*)(ws + 0);         // 4096x768      6,291,456 B
  __bf16* Wqt = (__bf16*)(ws + 6291456);   // 768x768 ^T    1,179,648 B
  __bf16* Wkt = (__bf16*)(ws + 7471104);
  __bf16* Wvt = (__bf16*)(ws + 8650752);
  __bf16* Wot = (__bf16*)(ws + 9830400);
  __bf16* Qbf = (__bf16*)(ws + 11010048);  // [b*H][S][64]  6,291,456 B
  __bf16* Kbf = (__bf16*)(ws + 17301504);
  __bf16* Vtb = (__bf16*)(ws + 23592960);  // [b*H][64][S]
  __bf16* Obf = (__bf16*)(ws + 29884416);  // [b][s][768]
  // total 36,175,872 B

  const int NX = ROWS * DMODEL;        // 3,145,728
  const int NW = DMODEL * DMODEL;      // 589,824
  cvt_f32_bf16<<<(NX + 255) / 256, 256, 0, stream>>>(x, Xbf, NX);
  transpose_f32_bf16<<<(NW + 255) / 256, 256, 0, stream>>>(Wq, Wqt, DMODEL, DMODEL);
  transpose_f32_bf16<<<(NW + 255) / 256, 256, 0, stream>>>(Wk, Wkt, DMODEL, DMODEL);
  transpose_f32_bf16<<<(NW + 255) / 256, 256, 0, stream>>>(Wv, Wvt, DMODEL, DMODEL);
  transpose_f32_bf16<<<(NW + 255) / 256, 256, 0, stream>>>(Wo, Wot, DMODEL, DMODEL);

  dim3 ggrid(ROWS / 64 / 4, DMODEL / 64);  // 16 x 12, 4 waves/block (64x64/wave)
  gemm_bf16_wmma<<<ggrid, 128, 0, stream>>>(Xbf, Wqt, bq, Qbf, nullptr,
                                            ROWS, DMODEL, DMODEL, 0);
  gemm_bf16_wmma<<<ggrid, 128, 0, stream>>>(Xbf, Wkt, bk, Kbf, nullptr,
                                            ROWS, DMODEL, DMODEL, 0);
  gemm_bf16_wmma<<<ggrid, 128, 0, stream>>>(Xbf, Wvt, bv, Vtb, nullptr,
                                            ROWS, DMODEL, DMODEL, 1);

  // 1536 32-query waves / 4 waves per block
  attn_wmma<<<384, 128, 0, stream>>>(Qbf, Kbf, Vtb, mk, Obf);

  gemm_bf16_wmma<<<ggrid, 128, 0, stream>>>(Obf, Wot, bo, nullptr, out,
                                            ROWS, DMODEL, DMODEL, 2);
}
